// Attention_40346922778795
// MI455X (gfx1250) — matrix operand order repaired and checked by reading, not compiled, not run on hardware
//
#include <hip/hip_runtime.h>

// ---------------------------------------------------------------------------
// Problem constants
// ---------------------------------------------------------------------------
#define BB   32
#define CC   128
#define HH   28
#define WW   28
#define LL   8
#define RR   32
#define HWp  784            // 28*28
#define CHW  100352         // 128*784
#define KK   1152           // C*9
#define MM   25088          // B*H*W
#define MT   1568           // MM/16

typedef __attribute__((ext_vector_type(16))) _Float16     v16h;
typedef __attribute__((ext_vector_type(8)))  float        v8f;
typedef __attribute__((ext_vector_type(8)))  unsigned int v8u;

__device__ __forceinline__ unsigned short f2h(float f) {
  _Float16 h = (_Float16)f;
  return __builtin_bit_cast(unsigned short, h);
}
__device__ __forceinline__ float sigf(float x) { return 1.0f / (1.0f + expf(-x)); }

__device__ __forceinline__ v16h load_frag16(const unsigned short* p0,
                                            const unsigned short* p1) {
  uint4 a = *reinterpret_cast<const uint4*>(p0);
  uint4 b = *reinterpret_cast<const uint4*>(p1);
  v8u u;
  u[0] = a.x; u[1] = a.y; u[2] = a.z; u[3] = a.w;
  u[4] = b.x; u[5] = b.y; u[6] = b.z; u[7] = b.w;
  return __builtin_bit_cast(v16h, u);
}

// ---------------------------------------------------------------------------
// fp32 -> fp16 conversion (weights)
// ---------------------------------------------------------------------------
__global__ void f2h_k(const float* __restrict__ in, unsigned short* __restrict__ out, int n) {
  int i = blockIdx.x * 256 + threadIdx.x;
  if (i < n) out[i] = f2h(in[i]);
}

// ---------------------------------------------------------------------------
// im2col: x [B,C,28,28] f32 -> A [M=25088, K=1152] f16, K = ci*9 + ky*3 + kx
// ---------------------------------------------------------------------------
__global__ void im2col_k(const float* __restrict__ x, unsigned short* __restrict__ A) {
  int i = blockIdx.x * 256 + threadIdx.x;
  if (i >= MM * KK) return;
  int k  = i % KK;
  int m  = i / KK;
  int ci = k / 9;
  int tap = k - ci * 9;
  int ky = tap / 3, kx = tap - ky * 3;
  int b  = m / HWp;
  int hw = m - b * HWp;
  int h  = hw / WW, w = hw - h * WW;
  int ih = h + ky - 1, iw = w + kx - 1;
  float val = 0.0f;
  if ((unsigned)ih < (unsigned)HH && (unsigned)iw < (unsigned)WW)
    val = x[((size_t)(b * CC + ci)) * HWp + ih * WW + iw];
  A[i] = f2h(val);
}

// ---------------------------------------------------------------------------
// WMMA implicit-GEMM conv: out[b][n][hw] = relu(A @ Bt^T + bias)
// A [M,K] f16 row-major, Bt [N=128,K] f16 row-major (== conv_w layer slice)
// Each wave: one 16(M) x 64(N) tile, 4 accumulators, K-loop of 36 steps.
// ---------------------------------------------------------------------------
__global__ void __launch_bounds__(256)
conv_gemm_k(const unsigned short* __restrict__ A,
            const unsigned short* __restrict__ Bt,
            const float* __restrict__ bias,
            float* __restrict__ out) {
  const int wave   = threadIdx.x >> 5;
  const int lane   = threadIdx.x & 31;
  const int gw     = blockIdx.x * 8 + wave;   // 0..3135
  const int m_tile = gw >> 1;                 // 0..1567
  const int n0     = (gw & 1) * 64;
  const int half   = lane >> 4;               // 0: lanes 0-15, 1: lanes 16-31
  const int lrow   = lane & 15;

  // A-frag per-lane base: row m, K offsets {half*8 + [0..7], +16}
  const unsigned short* arow = A + (size_t)(m_tile * 16 + lrow) * KK + half * 8;
  // B-frag per-lane bases: Bt row n, K offsets {half*8 + [0..7], +16}
  const unsigned short* brow[4];
#pragma unroll
  for (int j = 0; j < 4; ++j)
    brow[j] = Bt + (size_t)(n0 + j * 16 + lrow) * KK + half * 8;

  v8f acc[4] = {v8f{}, v8f{}, v8f{}, v8f{}};

  for (int k0 = 0; k0 < KK; k0 += 32) {
    v16h af = load_frag16(arow + k0, arow + k0 + 16);
#pragma unroll
    for (int j = 0; j < 4; ++j) {
      v16h bf = load_frag16(brow[j] + k0, brow[j] + k0 + 16);
      acc[j] = __builtin_amdgcn_wmma_f32_16x16x32_f16(
          /*neg_a=*/false, af, /*neg_b=*/false, bf,
          /*c_mod=*/(short)0, acc[j], /*reuse_a=*/false, /*reuse_b=*/false);
    }
  }

  // Epilogue: C/D layout -> lane<16: (M=r, N=lrow); lane>=16: (M=8+r, N=lrow)
  const int bb  = (m_tile * 16) / HWp;                 // batch (tile never crosses)
  const int hw0 = m_tile * 16 - bb * HWp + half * 8;
#pragma unroll
  for (int j = 0; j < 4; ++j) {
    const int n = n0 + j * 16 + lrow;
    const float bv = bias[n];
    float* ob = out + ((size_t)bb * CC + n) * HWp + hw0;
#pragma unroll
    for (int r = 0; r < 8; ++r) {
      float y = acc[j][r] + bv;
      ob[r] = fmaxf(y, 0.0f);
    }
  }
}

// ---------------------------------------------------------------------------
// Global average pool over 784 with optional compact->[slots] row remap:
// phys_row = r + (r / (t*128)) * (slots - t) * 128 ; out written at phys_row.
// ---------------------------------------------------------------------------
__global__ void gap_k(const float* __restrict__ in, float* __restrict__ out,
                      int nrows, int t, int slots) {
  int r = blockIdx.x * 8 + (threadIdx.x >> 5);
  if (r >= nrows) return;
  int lane = threadIdx.x & 31;
  int phys = r + (r / (t * CC)) * (slots - t) * CC;
  const float* p = in + (size_t)phys * HWp;
  float s = 0.0f;
  for (int i = lane; i < HWp; i += 32) s += p[i];
#pragma unroll
  for (int o = 16; o > 0; o >>= 1) s += __shfl_down(s, o, 32);
  if (lane == 0) out[phys] = s * (1.0f / (float)HWp);
}

// ---------------------------------------------------------------------------
// DSU cell: gates = cell_ih(x) + cell_hh(h); nc = sig(f)*cx + sig(i)*tanh(g);
// nh = sig(nc).  One block (128 thr) per logical row.
// phys_row = n + (n/t)*(slots-t).  hper/cper: 1 = per-row, 0 = broadcast row 0.
// ---------------------------------------------------------------------------
__global__ void __launch_bounds__(128)
dsu_k(const float* __restrict__ xin, int t, int slots,
      const float* __restrict__ hin, int hper,
      const float* __restrict__ cin, int cper,
      float* __restrict__ nh, float* __restrict__ nc,
      const float* __restrict__ ihw1, const float* __restrict__ ihb1,
      const float* __restrict__ ihw2, const float* __restrict__ ihb2,
      const float* __restrict__ hhw1, const float* __restrict__ hhb1,
      const float* __restrict__ hhw2, const float* __restrict__ hhb2) {
  const int n    = blockIdx.x;
  const int phys = n + (n / t) * (slots - t);
  const int tid  = threadIdx.x;
  __shared__ float xs[CC], hs[CC], z[2 * RR], g[3 * CC];

  xs[tid] = xin[(size_t)phys * CC + tid];
  hs[tid] = hin[(size_t)(hper ? n : 0) * CC + tid];
  __syncthreads();

  if (tid < 64) {
    const int r = tid & 31;
    const float* w1  = (tid < 32) ? ihw1 : hhw1;
    const float* b1  = (tid < 32) ? ihb1 : hhb1;
    const float* src = (tid < 32) ? xs : hs;
    float s = b1[r];
    for (int j = 0; j < CC; ++j) s += w1[r * CC + j] * src[j];
    z[tid] = fmaxf(s, 0.0f);
  }
  __syncthreads();

  for (int gi = tid; gi < 3 * CC; gi += 128) {
    float s = ihb2[gi] + hhb2[gi];
    const float* wi = ihw2 + gi * RR;
    const float* wh = hhw2 + gi * RR;
    for (int j = 0; j < RR; ++j) s += wi[j] * z[j] + wh[j] * z[RR + j];
    g[gi] = s;
  }
  __syncthreads();

  {
    const float ig = g[tid], fg = g[CC + tid], cg = g[2 * CC + tid];
    const float cx  = cin[(size_t)(cper ? n : 0) * CC + tid];
    const float ncv = sigf(fg) * cx + sigf(ig) * tanhf(cg);
    nh[(size_t)phys * CC + tid] = sigf(ncv);
    if (nc) nc[(size_t)phys * CC + tid] = ncv;
  }
}

// k *= kh over valid slots (compact index -> phys slot index)
__global__ void kmul_k(float* __restrict__ k, const float* __restrict__ kh,
                       int t, int total) {
  int i = blockIdx.x * 256 + threadIdx.x;
  if (i >= total) return;
  int b = i / (t * CC);
  int phys = b * LL * CC + (i - b * t * CC);
  k[phys] *= kh[phys];
}

// v[b][j][c][hw] *= vh[b][j][c]
__global__ void vmul_k(float* __restrict__ v, const float* __restrict__ vh, int t) {
  int i = blockIdx.x * 256 + threadIdx.x;
  int total = t * BB * CHW;
  if (i >= total) return;
  int per_b = t * CHW;
  int b = i / per_b;
  int rem = i - b * per_b;
  int j = rem / CHW;
  int cw = rem - j * CHW;
  int prow = b * LL + j;
  v[(size_t)prow * CHW + cw] *= vh[prow * CC + cw / HWp];
}

// softmax attention weights: w[b][j] = softmax_j(q[b].k[b][j]/sqrt(128))
__global__ void __launch_bounds__(128)
attw_k(const float* __restrict__ q, const float* __restrict__ k,
       float* __restrict__ w, int t) {
  const int b = blockIdx.x, tid = threadIdx.x;
  __shared__ float red[128];
  __shared__ float sc[LL];
  for (int j = 0; j < t; ++j) {
    red[tid] = q[b * CC + tid] * k[(size_t)(b * LL + j) * CC + tid];
    __syncthreads();
    for (int s = 64; s > 0; s >>= 1) {
      if (tid < s) red[tid] += red[tid + s];
      __syncthreads();
    }
    if (tid == 0) sc[j] = red[0] * 0.08838834764831845f;  // 1/sqrt(128)
    __syncthreads();
  }
  if (tid == 0) {
    float mx = -1e30f;
    for (int j = 0; j < t; ++j) mx = fmaxf(mx, sc[j]);
    float s = 0.0f;
    for (int j = 0; j < t; ++j) { float e = expf(sc[j] - mx); sc[j] = e; s += e; }
    float inv = 1.0f / s;
    for (int j = 0; j < t; ++j) w[b * LL + j] = sc[j] * inv;
  }
}

// x += sum_j w[b][j]*v[b][j]; append v[b][t] = x_new  (t==0: pure append)
__global__ void ctx_k(float* __restrict__ x, float* __restrict__ v,
                      const float* __restrict__ w, int t) {
  int i = blockIdx.x * 256 + threadIdx.x;
  if (i >= BB * CHW) return;
  int b = i / CHW;
  int rem = i - b * CHW;
  float s = x[i];
  const float* vb = v + (size_t)b * LL * CHW + rem;
  for (int j = 0; j < t; ++j) s += w[b * LL + j] * vb[(size_t)j * CHW];
  x[i] = s;
  v[(size_t)(b * LL + t) * CHW + rem] = s;
}

// append k[b][t] = q[b]
__global__ void kapp_k(float* __restrict__ k, const float* __restrict__ q, int t) {
  int i = blockIdx.x * 256 + threadIdx.x;
  if (i >= BB * CC) return;
  int b = i >> 7, c = i & (CC - 1);
  k[(size_t)(b * LL + t) * CC + c] = q[i];
}

// ---------------------------------------------------------------------------
// Host orchestration
// ---------------------------------------------------------------------------
extern "C" void kernel_launch(void* const* d_in, const int* in_sizes, int n_in,
                              void* d_out, int out_size, void* d_ws, size_t ws_size,
                              hipStream_t stream) {
  (void)in_sizes; (void)n_in; (void)ws_size;

  const float* X    = (const float*)d_in[0];
  const float* CW   = (const float*)d_in[1];
  const float* CB   = (const float*)d_in[2];
  const float* IHW1 = (const float*)d_in[3];
  const float* IHB1 = (const float*)d_in[4];
  const float* IHW2 = (const float*)d_in[5];
  const float* IHB2 = (const float*)d_in[6];
  const float* HHW1 = (const float*)d_in[7];
  const float* HHB1 = (const float*)d_in[8];
  const float* HHW2 = (const float*)d_in[9];
  const float* HHB2 = (const float*)d_in[10];

  // Workspace layout (all offsets 256B aligned); total ~189.2 MB
  char* base = (char*)d_ws;
  size_t off = 0;
  unsigned short* Abf = (unsigned short*)(base + off); off += (size_t)MM * KK * 2;       // 57,802,752
  unsigned short* Wbf = (unsigned short*)(base + off); off += (size_t)LL * CC * KK * 2;  //  2,359,296
  float* xb0    = (float*)(base + off); off += (size_t)BB * CHW * 4;                     // 12,845,056
  float* xb1    = (float*)(base + off); off += (size_t)BB * CHW * 4;
  float* vbuf   = (float*)(base + off); off += (size_t)BB * LL * CHW * 4;                // 102,760,448
  float* kbuf   = (float*)(base + off); off += (size_t)BB * LL * CC * 4;
  float* khbuf  = (float*)(base + off); off += (size_t)BB * LL * CC * 4;
  float* pooled = (float*)(base + off); off += (size_t)BB * LL * CC * 4;
  float* vhbuf  = (float*)(base + off); off += (size_t)BB * LL * CC * 4;
  float* seq    = (float*)(base + off); off += (size_t)BB * CC * 4;
  float* ht0    = (float*)(base + off); off += (size_t)BB * CC * 4;
  float* ct0    = (float*)(base + off); off += (size_t)BB * CC * 4;
  float* ht1    = (float*)(base + off); off += (size_t)BB * CC * 4;
  float* ct1    = (float*)(base + off); off += (size_t)BB * CC * 4;
  float* attw   = (float*)(base + off); off += 1024;

  float* xb[2] = {xb0, xb1};

  // zero initial hidden/cell state (capture-safe)
  hipMemsetAsync(ht0, 0, (size_t)BB * CC * 4, stream);
  hipMemsetAsync(ct0, 0, (size_t)BB * CC * 4, stream);

  // convert all conv weights to f16 once (Bt layout == conv_w layout)
  {
    int n = LL * CC * KK;  // 1,179,648
    f2h_k<<<(n + 255) / 256, 256, 0, stream>>>(CW, Wbf, n);
  }

  float *hA = ht0, *cA = ct0, *hB = ht1, *cB = ct1;

  for (int idx = 0; idx < LL; ++idx) {
    const float* xsrc = (idx == 0) ? X : xb[(idx - 1) & 1];
    float* xdst = xb[idx & 1];

    // conv + bias + relu via WMMA implicit GEMM
    im2col_k<<<(MM * KK + 255) / 256, 256, 0, stream>>>(xsrc, Abf);
    conv_gemm_k<<<MT * 2 / 8, 256, 0, stream>>>(Abf, Wbf + (size_t)idx * CC * KK,
                                                CB + idx * CC, xdst);

    // seq (== attention q) = GAP(x)
    gap_k<<<(BB * CC + 7) / 8, 256, 0, stream>>>(xdst, seq, BB * CC, 1, 1);

    // DSU state update (per-row h,c); double-buffered
    dsu_k<<<BB, 128, 0, stream>>>(seq, 1, 1, hA, 1, cA, 1, hB, cB,
                                  IHW1, IHB1, IHW2, IHB2, HHW1, HHB1, HHW2, HHB2);
    { float* tp = hA; hA = hB; hB = tp; tp = cA; cA = cB; cB = tp; }

    const int t = idx;
    if (t > 0) {
      // k modulation: dsu over k rows, h/c broadcast from batch 0 of NEW state
      dsu_k<<<BB * t, 128, 0, stream>>>(kbuf, t, LL, hA, 0, cA, 0, khbuf, nullptr,
                                        IHW1, IHB1, IHW2, IHB2, HHW1, HHB1, HHW2, HHB2);
      kmul_k<<<(BB * t * CC + 255) / 256, 256, 0, stream>>>(kbuf, khbuf, t, BB * t * CC);

      // v modulation via GAP of stored maps
      gap_k<<<(BB * t * CC + 7) / 8, 256, 0, stream>>>(vbuf, pooled, BB * t * CC, t, LL);
      dsu_k<<<BB * t, 128, 0, stream>>>(pooled, t, LL, hA, 0, cA, 0, vhbuf, nullptr,
                                        IHW1, IHB1, IHW2, IHB2, HHW1, HHB1, HHW2, HHB2);
      vmul_k<<<(BB * t * CHW + 255) / 256, 256, 0, stream>>>(vbuf, vhbuf, t);

      // attention weights over modulated k
      attw_k<<<BB, 128, 0, stream>>>(seq, kbuf, attw, t);
    }

    // x += ctx (noop at t==0); append v[t] = x_new; append k[t] = q
    ctx_k<<<(BB * CHW + 255) / 256, 256, 0, stream>>>(xdst, vbuf, attw, t);
    kapp_k<<<(BB * CC + 255) / 256, 256, 0, stream>>>(kbuf, seq, t);
  }

  hipMemcpyAsync(d_out, xb[(LL - 1) & 1], (size_t)out_size * sizeof(float),
                 hipMemcpyDeviceToDevice, stream);
}
